// SingleHeadAttention_85547158602294
// MI455X (gfx1250) — matrix operand order repaired and checked by reading, not compiled, not run on hardware
//
#include <hip/hip_runtime.h>
#include <hip/hip_bf16.h>

typedef __attribute__((ext_vector_type(16))) _Float16 v16h;
typedef __attribute__((ext_vector_type(8)))  float    v8f;

#define BATCH 4
#define SEQ   4096
#define EMB   1024
#define HD    64

static __device__ __forceinline__ v8f wmma_f16(v16h a, v16h b, v8f c) {
  // D = A(16x32 f16) * B(32x16 f16) + C(16x16 f32)
  return __builtin_amdgcn_wmma_f32_16x16x32_f16(false, a, false, b, (short)0, c,
                                                false, false);
}

// ---------------------------------------------------------------------------
// Projection: K/Q/V = embedded @ W^T, fp32 in -> f16 out.
// Block = 12 waves: wave w -> weight (w>>2 : 0=K,1=Q,2=V), col tile (w&3).
// Each wave computes one 16x16 output tile with 32 WMMAs over K=1024.
// V is stored transposed [B][64][S] for the PV B-operand.
// ---------------------------------------------------------------------------
__global__ __launch_bounds__(384)
void proj_kernel(const float* __restrict__ emb,
                 const float* __restrict__ Wk,
                 const float* __restrict__ Wq,
                 const float* __restrict__ Wv,
                 _Float16* __restrict__ Kh,
                 _Float16* __restrict__ Qh,
                 _Float16* __restrict__ Vt) {
  const int lane = threadIdx.x & 31;
  const int wv   = threadIdx.x >> 5;   // 0..11
  const int wsel = wv >> 2;            // 0=K, 1=Q, 2=V
  const int ct   = wv & 3;             // 16-wide column tile of the 64 outputs
  const int hb   = lane >> 4;          // lane half (0/1)
  const int lm   = lane & 15;
  const int s0   = blockIdx.x * 16;
  const int b    = blockIdx.y;

  const float* Wm   = (wsel == 0) ? Wk : (wsel == 1) ? Wq : Wv;
  const float* arow = emb + ((size_t)b * SEQ + s0 + lm) * EMB;     // A: row lm
  const float* brow = Wm + (size_t)(ct * 16 + lm) * EMB;           // B: col lm

  v8f acc = {};
  for (int e0 = 0; e0 < EMB; e0 += 32) {
    v16h av, bvv;
    const float* pa = arow + e0 + hb * 16;
    const float* pb = brow + e0 + hb * 16;
#pragma unroll
    for (int i = 0; i < 8; ++i) {
      av[i]     = (_Float16)pa[i];        // K = hb*16 + i
      av[8 + i] = (_Float16)pa[8 + i];    // K = hb*16 + 8 + i
    }
#pragma unroll
    for (int i = 0; i < 16; ++i) bvv[i] = (_Float16)pb[i];  // K = hb*16 + i
    acc = wmma_f16(av, bvv, acc);
  }

  // C/D layout: VGPR v -> row M = hb*8+v, col N = lm
  if (wsel == 2) {
#pragma unroll
    for (int v = 0; v < 8; ++v)
      Vt[((size_t)b * HD + ct * 16 + lm) * SEQ + s0 + hb * 8 + v] =
          (_Float16)acc[v];
  } else {
    _Float16* dst = (wsel == 0) ? Kh : Qh;
#pragma unroll
    for (int v = 0; v < 8; ++v)
      dst[((size_t)b * SEQ + s0 + hb * 8 + v) * HD + ct * 16 + lm] =
          (_Float16)acc[v];
  }
}

// ---------------------------------------------------------------------------
// Flash attention: one wave per 16-query tile, causal, online softmax.
// K tiles (32 keys x 64 f16 = 4KB, contiguous in Kh) are double-buffered into
// LDS with GLOBAL_LOAD_ASYNC_TO_LDS_B128 (ASYNCcnt), issued one iteration
// ahead. Per chunk: 4 score WMMAs, mask + online softmax in C/D layout
// (shfl_xor row reductions in 16-lane halves), P relayout via LDS, 4 PV WMMAs.
// ---------------------------------------------------------------------------
#define KPAD 72   // halves per row: 144B row stride, 16B aligned, 36*r%64 perm
#define PPAD 40   // halves per row:  80B row stride, 16B aligned, 20*r%64 perm

__global__ __launch_bounds__(32)
void attn_kernel(const _Float16* __restrict__ Kh,
                 const _Float16* __restrict__ Qh,
                 const _Float16* __restrict__ Vt,
                 float* __restrict__ out) {
  __shared__ __align__(16) _Float16 kbuf[2][32][KPAD];  // double-buffered K
  __shared__ __align__(16) _Float16 pt[16][PPAD];       // P transpose tile

  const int lane = threadIdx.x & 31;
  const int hb   = lane >> 4;
  const int lm   = lane & 15;
  const int q0   = blockIdx.x * 16;
  const int b    = blockIdx.y;

  // Per-lane async-copy addressing: lane covers 16B of each 512B global /
  // 576B padded-LDS slab; 8 issues move the whole 4KB tile.
  const uint32_t lds_lane_off =
      (uint32_t)((lane >> 3) * (KPAD * 2) + (lane & 7) * 16);
  const uint32_t kbase[2] = {(uint32_t)(size_t)(&kbuf[0][0][0]),
                             (uint32_t)(size_t)(&kbuf[1][0][0])};

  // Q A-operands (head dims 0..31 and 32..63), scaled by 1/sqrt(64)=0.125.
  const _Float16* qrow = Qh + ((size_t)b * SEQ + q0 + lm) * HD;
  const _Float16 scl = (_Float16)0.125f;
  v16h aq0, aq1;
#pragma unroll
  for (int i = 0; i < 8; ++i) {
    aq0[i]     = qrow[hb * 16 + i] * scl;
    aq0[8 + i] = qrow[hb * 16 + 8 + i] * scl;
    aq1[i]     = qrow[32 + hb * 16 + i] * scl;
    aq1[8 + i] = qrow[32 + hb * 16 + 8 + i] * scl;
  }

  float mr[8], lr[8];
  v8f zf = {};
  v8f acc[4];
#pragma unroll
  for (int v = 0; v < 8; ++v) { mr[v] = -1e30f; lr[v] = 0.f; }
#pragma unroll
  for (int c = 0; c < 4; ++c) acc[c] = zf;

  const int nkt = (q0 + 15) / 32 + 1;   // causal: key chunks up to diagonal

  // Issue async copy of K tile `kt` into buffer `bi`.
  auto issue_ktile = [&](int bi, int kt) {
    uint32_t ldsa = kbase[bi] + lds_lane_off;
    const char* ga =
        (const char*)(Kh + ((size_t)b * SEQ + kt * 32) * HD) + lane * 16;
#pragma unroll
    for (int j = 0; j < 8; ++j) {
      asm volatile("global_load_async_to_lds_b128 %0, %1, off"
                   :: "v"(ldsa), "v"(ga)
                   : "memory");
      ldsa += (uint32_t)(4 * KPAD * 2);  // 4 padded rows
      ga   += 512;                       // 4 dense rows
    }
  };

  issue_ktile(0, 0);  // prologue

  for (int kt = 0; kt < nkt; ++kt) {
    const int k0  = kt * 32;
    const int cur = kt & 1;

    if (kt + 1 < nkt) {
      issue_ktile(cur ^ 1, kt + 1);
      // Async loads complete in issue order: <=8 outstanding => tile kt done.
      asm volatile("s_wait_asynccnt 0x8" ::: "memory");
    } else {
      asm volatile("s_wait_asynccnt 0x0" ::: "memory");
    }

    // K B-operands from LDS: lane -> key column lm (tile0) / 16+lm (tile1).
    v16h kb00, kb01, kb10, kb11;
#pragma unroll
    for (int i = 0; i < 16; ++i) {
      kb00[i] = kbuf[cur][lm][hb * 16 + i];
      kb01[i] = kbuf[cur][lm][32 + hb * 16 + i];
      kb10[i] = kbuf[cur][16 + lm][hb * 16 + i];
      kb11[i] = kbuf[cur][16 + lm][32 + hb * 16 + i];
    }
    v8f sc0 = zf, sc1 = zf;
    sc0 = wmma_f16(aq0, kb00, sc0);
    sc0 = wmma_f16(aq1, kb01, sc0);
    sc1 = wmma_f16(aq0, kb10, sc1);
    sc1 = wmma_f16(aq1, kb11, sc1);

    float p0v[8], p1v[8];
#pragma unroll
    for (int v = 0; v < 8; ++v) {
      const int qg = q0 + hb * 8 + v;
      float x0 = (k0 + lm      <= qg) ? sc0[v] : -1e30f;
      float x1 = (k0 + 16 + lm <= qg) ? sc1[v] : -1e30f;
      float xm = fmaxf(x0, x1);
#pragma unroll
      for (int m = 1; m <= 8; m <<= 1) xm = fmaxf(xm, __shfl_xor(xm, m, 32));
      const float mn = fmaxf(mr[v], xm);
      const float p0 = __expf(x0 - mn);
      const float p1 = __expf(x1 - mn);
      float rs = p0 + p1;
#pragma unroll
      for (int m = 1; m <= 8; m <<= 1) rs += __shfl_xor(rs, m, 32);
      const float al = __expf(mr[v] - mn);
      lr[v] = lr[v] * al + rs;
      mr[v] = mn;
#pragma unroll
      for (int c = 0; c < 4; ++c) acc[c][v] *= al;
      p0v[v] = p0;
      p1v[v] = p1;
    }

    // C/D layout -> LDS -> A-operand layout for the PV WMMA.
#pragma unroll
    for (int v = 0; v < 8; ++v) {
      pt[hb * 8 + v][lm]      = (_Float16)p0v[v];
      pt[hb * 8 + v][16 + lm] = (_Float16)p1v[v];
    }
    __syncthreads();
    v16h ap;
#pragma unroll
    for (int i = 0; i < 8; ++i) {
      ap[i]     = pt[lm][hb * 16 + i];
      ap[8 + i] = pt[lm][hb * 16 + 8 + i];
    }
    __syncthreads();

#pragma unroll
    for (int c = 0; c < 4; ++c) {
      const _Float16* vp =
          Vt + ((size_t)b * HD + c * 16 + lm) * SEQ + k0 + hb * 16;
      v16h bv;
#pragma unroll
      for (int i = 0; i < 16; ++i) bv[i] = vp[i];
      acc[c] = wmma_f16(ap, bv, acc[c]);
    }
  }

  // Normalize, round to 4 decimals (reference does jnp.round(..., 4)), store.
#pragma unroll
  for (int c = 0; c < 4; ++c)
#pragma unroll
    for (int v = 0; v < 8; ++v) {
      const float val = acc[c][v] / lr[v];
      out[((size_t)b * SEQ + q0 + hb * 8 + v) * HD + c * 16 + lm] =
          rintf(val * 1e4f) * 1e-4f;
    }
}

extern "C" void kernel_launch(void* const* d_in, const int* in_sizes, int n_in,
                              void* d_out, int out_size, void* d_ws,
                              size_t ws_size, hipStream_t stream) {
  const float* emb = (const float*)d_in[0];
  const float* Wk  = (const float*)d_in[1];
  const float* Wq  = (const float*)d_in[2];
  const float* Wv  = (const float*)d_in[3];
  float* out = (float*)d_out;

  const size_t per = (size_t)BATCH * SEQ * HD;  // 1 Mi elems (2 MB f16 each)
  _Float16* Kh = (_Float16*)d_ws;
  _Float16* Qh = Kh + per;
  _Float16* Vt = Qh + per;

  proj_kernel<<<dim3(SEQ / 16, BATCH), 384, 0, stream>>>(emb, Wk, Wq, Wv, Kh,
                                                         Qh, Vt);
  attn_kernel<<<dim3(SEQ / 16, BATCH), 32, 0, stream>>>(Kh, Qh, Vt, out);
}
